// loss_PDE_layer_60851096649857
// MI455X (gfx1250) — compile-verified, matrix operand order checked by reading, not run on hardware
//
#include <hip/hip_runtime.h>
#include <hip/hip_fp16.h>

typedef __attribute__((ext_vector_type(16))) _Float16 v16h;
typedef __attribute__((ext_vector_type(8)))  float    v8f;

#define N_PTS  262144
#define HIDDEN 200
#define KTILES 13          // ceil(200/16) hidden-unit tiles of 16
#define WAVES_PER_BLOCK 8  // 256 threads, wave32
#define PTS_PER_WAVE 16

__device__ __forceinline__ float fast_tanh(float z) {
#if __has_builtin(__builtin_amdgcn_tanhf)
    return __builtin_amdgcn_tanhf(z);          // v_tanh_f32 (gfx1250 TRANS)
#else
    float e = __expf(2.0f * z);
    return 1.0f - 2.0f * __builtin_amdgcn_rcpf(e + 1.0f);
#endif
}

__global__ void zero_out_kernel(float* out) {
    if (threadIdx.x == 0) out[0] = 0.0f;
}

__global__ __launch_bounds__(256) void pinn_lap_loss_kernel(
    const float* __restrict__ x,  const float* __restrict__ y,
    const float* __restrict__ W1, const float* __restrict__ b1,
    const float* __restrict__ W2, float* __restrict__ out)
{
    const int lane = threadIdx.x & 31;
    const int wave = threadIdx.x >> 5;
    const int m    = lane & 15;

    // ---------------------------------------------------------------
    // Once-per-block prep: pack B fragments (f16) + d factors into LDS.
    // B fragment per lane only needs elems 0..3 (K=0..2 used, K=3 pad):
    //   dword0 = {W1[0,j], W1[1,j]}  dword1 = {b1[j], 0}  (lanes 0..15)
    //   dword0 = dword1 = 0                               (lanes 16..31)
    // d[j] = -2 * W2[j] * (W1[0,j]^2 + W1[1,j]^2)
    // ---------------------------------------------------------------
    __shared__ uint2 sB[KTILES][32];
    __shared__ float sD[KTILES][16];
    __shared__ float bsum;

    for (int idx = threadIdx.x; idx < KTILES * 32; idx += 256) {
        const int kt = idx >> 5;
        const int ln = idx & 31;
        const int mm = ln & 15;
        const int j  = kt * 16 + mm;
        const float v  = (j < HIDDEN) ? 1.0f : 0.0f;      // pad mask
        const int   jc = (j < HIDDEN) ? j : (HIDDEN - 1); // clamp (in-bounds)

        const float w0 = W1[jc]          * v;   // W1[0, j]
        const float w1 = W1[HIDDEN + jc] * v;   // W1[1, j]
        const float bb = b1[jc]          * v;
        const float w2 = W2[jc]          * v;

        const bool lo = (ln < 16);
        union { unsigned u; _Float16 h[2]; } p0, p1;
        p0.h[0] = lo ? (_Float16)w0 : (_Float16)0.0f;
        p0.h[1] = lo ? (_Float16)w1 : (_Float16)0.0f;
        p1.h[0] = lo ? (_Float16)bb : (_Float16)0.0f;
        p1.h[1] = (_Float16)0.0f;
        sB[kt][ln] = make_uint2(p0.u, p1.u);
        if (lo) sD[kt][mm] = -2.0f * w2 * (w0 * w0 + w1 * w1);
    }
    if (threadIdx.x == 0) bsum = 0.0f;
    __syncthreads();

    // ---------------- A fragment: 16x32 f16 (built once) ----------------
    // Lane m (0..15), row M=m: elem0 = x_m, elem1 = y_m, elem2 = 1 (bias).
    // Lanes 16..31 hold K=8..15, 24..31 -> all zero.
    const int waveGlobal = blockIdx.x * WAVES_PER_BLOCK + wave;
    const int ptBase = waveGlobal * PTS_PER_WAVE;   // 16 points per wave
    const float px = x[ptBase + m];
    const float py = y[ptBase + m];
    v16h afrag = {};
    if (lane < 16) {
        afrag[0] = (_Float16)px;
        afrag[1] = (_Float16)py;
        afrag[2] = (_Float16)1.0f;
    }

    float acc[8];
#pragma unroll
    for (int r = 0; r < 8; ++r) acc[r] = 0.0f;

    // ---------------- Hot loop: 13 hidden-unit tiles ----------------
    for (int kt = 0; kt < KTILES; ++kt) {
        const uint2 bw = sB[kt][lane];   // ds_load_b64
        const float d  = sD[kt][m];      // ds_load_b32

        union { unsigned u; _Float16 h[2]; } c0, c1;
        c0.u = bw.x; c1.u = bw.y;
        v16h bfrag = {};
        bfrag[0] = c0.h[0];
        bfrag[1] = c0.h[1];
        bfrag[2] = c1.h[0];
        bfrag[3] = c1.h[1];

        // z-tile = A x B: 16 points x 16 hidden units
        v8f c = {};
        c = __builtin_amdgcn_wmma_f32_16x16x32_f16(
                /*neg_a=*/false, afrag, /*neg_b=*/false, bfrag,
                /*c_mod=*/(short)0, c, /*reuse_a=*/false, /*reuse_b=*/false);

        // acc_r += d_j * (t - t^3),  t = tanh(z)
#pragma unroll
        for (int r = 0; r < 8; ++r) {
            float t = fast_tanh(c[r]);
            float s = t * t;
            float u = fmaf(-s, t, t);       // t * (1 - t^2)
            acc[r]  = fmaf(d, u, acc[r]);
        }
    }

    // XOR-butterfly over the 16 hidden-unit lanes of each half-wave.
    // Lanes [0..15] end with lap for rows 0..7, lanes [16..31] rows 8..15.
#pragma unroll
    for (int r = 0; r < 8; ++r) {
        float vsum = acc[r];
        vsum += __shfl_xor(vsum, 1, 32);
        vsum += __shfl_xor(vsum, 2, 32);
        vsum += __shfl_xor(vsum, 4, 32);
        vsum += __shfl_xor(vsum, 8, 32);
        acc[r] = vsum;
    }

    // Lanes 0 and 16 finish their 8 rows: residual vs rhs, sum of squares.
    float ssum = 0.0f;
    if ((lane & 15) == 0) {
        const int rowBase = ptBase + (lane >> 4) * 8;
#pragma unroll
        for (int r = 0; r < 8; ++r) {
            float xx  = x[rowBase + r];
            float yy  = y[rowBase + r];
            float rhs = -8.0f * __sinf(2.0f * xx) * __cosf(2.0f * yy);
            float e   = acc[r] - rhs;
            ssum = fmaf(e, e, ssum);
        }
    }

    // Block-level accumulation, then one global atomic per block.
    __syncthreads();   // reuse bsum safely after prep init
    if ((lane & 15) == 0) atomicAdd(&bsum, ssum);
    __syncthreads();
    if (threadIdx.x == 0) atomicAdd(out, bsum * (1.0f / (float)N_PTS));
}

extern "C" void kernel_launch(void* const* d_in, const int* in_sizes, int n_in,
                              void* d_out, int out_size, void* d_ws, size_t ws_size,
                              hipStream_t stream) {
    const float* x  = (const float*)d_in[0];
    const float* y  = (const float*)d_in[1];
    const float* W1 = (const float*)d_in[2];  // (2, 200) row-major
    const float* b1 = (const float*)d_in[3];  // (200,)
    const float* W2 = (const float*)d_in[4];  // (200, 1)
    // d_in[5] = b2: constant offset, vanishes under the Laplacian.
    float* out = (float*)d_out;

    zero_out_kernel<<<1, 64, 0, stream>>>(out);

    const int blocks = N_PTS / (WAVES_PER_BLOCK * PTS_PER_WAVE);  // 2048
    pinn_lap_loss_kernel<<<blocks, 256, 0, stream>>>(x, y, W1, b1, W2, out);
}
